// GAT_7267084665079
// MI455X (gfx1250) — compile-verified, matrix operand order checked by reading, not run on hardware
//
#include <hip/hip_runtime.h>
#include <cmath>

typedef __attribute__((ext_vector_type(16))) _Float16 v16h;
typedef __attribute__((ext_vector_type(8)))  float    v8f;

static inline unsigned cdiv_u(long long a, long long b) { return (unsigned)((a + b - 1) / b); }

// ----------------------------- utility kernels -----------------------------

__global__ void k_fill_f32(float* __restrict__ p, float v, long long n) {
  long long i = (long long)blockIdx.x * blockDim.x + threadIdx.x;
  if (i < n) p[i] = v;
}

__global__ void k_f32_to_f16(const float* __restrict__ s, _Float16* __restrict__ d, long long n) {
  long long i = (long long)blockIdx.x * blockDim.x + threadIdx.x;
  if (i < n) d[i] = (_Float16)s[i];
}

// Wt[n*K + k] = W[k*Nout + n]  (weights K-contiguous per output column)
__global__ void k_transpose_f16(const float* __restrict__ W, _Float16* __restrict__ Wt,
                                int K, int Nout) {
  int i = blockIdx.x * blockDim.x + threadIdx.x;
  if (i < K * Nout) {
    int k = i / Nout, n = i % Nout;
    Wt[(long long)n * K + k] = (_Float16)W[i];
  }
}

// ----------------------------- WMMA GEMM -----------------------------------
// C[N,NOUT] = A[N,K] (f16 row-major) x Bt[NOUT,K] (f16 K-contiguous), f32 accum.
// One wave owns a 16x64 strip of C (4 accumulators, A fragment reused 4x).
// K and NOUT are compile-time: only 2 base pointers, all fragment loads use
// immediate offsets, k-loop fully unrolled -> no induction math, no spills.
// Requires N%16==0 (50000=3125*16), K%32==0, NOUT%64==0.
struct Frag { union { v16h v; unsigned u[8]; }; };

__device__ __forceinline__ void mma4(const Frag& a, const Frag (&b)[4], v8f (&acc)[4]) {
#pragma unroll
  for (int s = 0; s < 4; ++s)
    acc[s] = __builtin_amdgcn_wmma_f32_16x16x32_f16(
        /*neg_a=*/false, a.v, /*neg_b=*/false, b[s].v,
        /*c_mod=*/(short)0, acc[s], /*reuse_a=*/false, /*reuse_b=*/false);
}

template <int K, int NOUT>
__global__ void k_wmma_gemm4(const _Float16* __restrict__ A, const _Float16* __restrict__ Bt,
                             float* __restrict__ C, int totalGroups) {
  constexpr int groupsN = NOUT / 64;
  constexpr int Kd = K / 2;              // dwords per f16 row
  int wid = (int)(((long long)blockIdx.x * blockDim.x + threadIdx.x) >> 5);
  if (wid >= totalGroups) return;        // uniform per wave -> EXEC stays all-ones
  int lane = threadIdx.x & 31;
  int tm = wid / groupsN, gn = wid % groupsN;
  int l16 = lane & 15, half = lane >> 4;

  // Pre-bias both base pointers by the half-dependent lane offset (4 dwords),
  // so all loads below are pure immediate offsets.
  const unsigned* __restrict__ Ar =
      (const unsigned*)(A + (long long)(tm * 16 + l16) * K) + 4 * half;
  const unsigned* __restrict__ Bp =
      (const unsigned*)(Bt + (long long)(gn * 64 + l16) * K) + 4 * half;

  v8f acc[4] = {{}, {}, {}, {}};
#pragma unroll
  for (int kb = 0; kb < K; kb += 32) {
    const int base = kb >> 1;
    Frag ua, ub[4];
    // 16-bit A-matrix 16x32 layout (ptr pre-biased by 4*half dwords):
    //   q 0..3 -> dword base+q ; q 4..7 -> dword base+8+(q-4)
#pragma unroll
    for (int q = 0; q < 8; ++q) {
      const int di = base + ((q < 4) ? 0 : 8) + (q & 3);
      ua.u[q] = Ar[di];
#pragma unroll
      for (int s = 0; s < 4; ++s) ub[s].u[q] = Bp[s * (16 * Kd) + di];
    }
    mma4(ua, ub, acc);
  }

  // C/D layout: lane l16 = column, VGPR r -> row r + 8*half
  int mbase = tm * 16 + 8 * half;
#pragma unroll
  for (int s = 0; s < 4; ++s) {
    int col = gn * 64 + s * 16 + l16;
#pragma unroll
    for (int r = 0; r < 8; ++r)
      C[(long long)(mbase + r) * NOUT + col] = acc[s][r];
  }
}

// ----------------------------- attention kernels ---------------------------

__device__ inline float leaky02(float x) { return x > 0.f ? x : 0.2f * x; }

__device__ inline void atomicMaxF(float* addr, float val) {
  if (val >= 0.f) atomicMax((int*)addr, __float_as_int(val));
  else            atomicMin((unsigned int*)addr, __float_as_uint(val));
}

// alpha_s[n,h] = sum_c H[n,h,c]*a_s[h,c]; same for alpha_d
__global__ void k_alpha(const float* __restrict__ H, const float* __restrict__ avs,
                        const float* __restrict__ avd, float* __restrict__ as_,
                        float* __restrict__ ad_, int N, int NH, int C) {
  int i = blockIdx.x * blockDim.x + threadIdx.x;
  if (i >= N * NH) return;
  int n = i / NH, h = i % NH;
  const float* hp = H + (long long)n * NH * C + (long long)h * C;
  const float* vs = avs + h * C;
  const float* vd = avd + h * C;
  float ss = 0.f, sd = 0.f;
  for (int c = 0; c < C; ++c) { float x = hp[c]; ss += x * vs[c]; sd += x * vd[c]; }
  as_[i] = ss;
  ad_[i] = sd;
}

__device__ inline void edge_sd(const int* __restrict__ ei, long long e, int E, int N,
                               int& s, int& d) {
  if (e < E) { s = ei[e]; d = ei[(long long)E + e]; }
  else       { s = d = (int)(e - E); }
}

// pass 1: segment max of leaky(alpha_s[src]+alpha_d[dst]) over dst
__global__ void k_edge_max(const int* __restrict__ ei, int E, int N,
                           const float* __restrict__ as_, const float* __restrict__ ad_,
                           float* __restrict__ m_, int NH) {
  long long i = (long long)blockIdx.x * blockDim.x + threadIdx.x;
  long long tot = (long long)(E + N) * NH;
  if (i >= tot) return;
  int h = (int)(i % NH);
  long long e = i / NH;
  int s, d; edge_sd(ei, e, E, N, s, d);
  float lg = leaky02(as_[(long long)s * NH + h] + ad_[(long long)d * NH + h]);
  atomicMaxF(&m_[(long long)d * NH + h], lg);
}

// pass 2: den[dst,h] += exp(logit - m[dst,h])
__global__ void k_edge_den(const int* __restrict__ ei, int E, int N,
                           const float* __restrict__ as_, const float* __restrict__ ad_,
                           const float* __restrict__ m_, float* __restrict__ den_, int NH) {
  long long i = (long long)blockIdx.x * blockDim.x + threadIdx.x;
  long long tot = (long long)(E + N) * NH;
  if (i >= tot) return;
  int h = (int)(i % NH);
  long long e = i / NH;
  int s, d; edge_sd(ei, e, E, N, s, d);
  long long nh = (long long)d * NH + h;
  float lg = leaky02(as_[(long long)s * NH + h] + ad_[nh]);
  atomicAdd(&den_[nh], expf(lg - m_[nh]));
}

// pass 3: materialize normalized attention weight per (edge, head)
__global__ void k_edge_alpha(const int* __restrict__ ei, int E, int N,
                             const float* __restrict__ as_, const float* __restrict__ ad_,
                             const float* __restrict__ m_, const float* __restrict__ den_,
                             float* __restrict__ alpha_e, int NH) {
  long long i = (long long)blockIdx.x * blockDim.x + threadIdx.x;
  long long tot = (long long)(E + N) * NH;
  if (i >= tot) return;
  int h = (int)(i % NH);
  long long e = i / NH;
  int s, d; edge_sd(ei, e, E, N, s, d);
  long long nh = (long long)d * NH + h;
  float lg = leaky02(as_[(long long)s * NH + h] + ad_[nh]);
  alpha_e[i] = expf(lg - m_[nh]) / (den_[nh] + 1e-16f);
}

// pass 4: Out[dst, f] += H[src, f] * alpha ; float4 per thread
__global__ void k_edge_acc4(const int* __restrict__ ei, int E, int N,
                            const float* __restrict__ alpha_e,
                            const float* __restrict__ H, float* __restrict__ Out,
                            int NH, int C) {
  int HC = NH * C, HC4 = HC >> 2;
  long long i = (long long)blockIdx.x * blockDim.x + threadIdx.x;
  long long tot = (long long)(E + N) * HC4;
  if (i >= tot) return;
  int f4 = (int)(i % HC4);
  long long e = i / HC4;
  int h = (f4 * 4) / C;
  int s, d; edge_sd(ei, e, E, N, s, d);
  float al = alpha_e[e * NH + h];
  const float4 hv = *(const float4*)(H + (long long)s * HC + f4 * 4);
  float* op = Out + (long long)d * HC + f4 * 4;
  atomicAdd(op + 0, hv.x * al);
  atomicAdd(op + 1, hv.y * al);
  atomicAdd(op + 2, hv.z * al);
  atomicAdd(op + 3, hv.w * al);
}

__global__ void k_bias_act(float* __restrict__ O, const float* __restrict__ b,
                           long long n, int HC, int do_elu) {
  long long i = (long long)blockIdx.x * blockDim.x + threadIdx.x;
  if (i >= n) return;
  float x = O[i] + b[(int)(i % HC)];
  if (do_elu) x = x > 0.f ? x : (expf(x) - 1.f);
  O[i] = x;
}

// ----------------------------- pooling / head ------------------------------

__global__ void k_pool(const float* __restrict__ X, const int* __restrict__ batch,
                       float* __restrict__ pooled, float* __restrict__ counts,
                       int N, int F) {
  int i = blockIdx.x * blockDim.x + threadIdx.x;
  if (i >= N * F) return;
  int n = i / F, f = i % F;
  int b = batch[n];
  atomicAdd(&pooled[(long long)b * F + f], X[i]);
  if (f == 0) atomicAdd(&counts[b], 1.f);
}

__global__ void k_final(const float* __restrict__ pooled, const float* __restrict__ counts,
                        const float* __restrict__ Wl, const float* __restrict__ bl,
                        float* __restrict__ out, int B, int F) {
  __shared__ float mx;
  if (threadIdx.x == 0) {
    float mm = 0.f;
    for (int b = 0; b < B; ++b) mm = fmaxf(mm, counts[b]);
    mx = mm;
  }
  __syncthreads();
  int b = threadIdx.x;
  if (b < B) {
    float acc = 0.f;
    for (int f = 0; f < F; ++f) acc += pooled[(long long)b * F + f] * Wl[f];
    out[b] = acc / mx + bl[0];
  }
}

// ----------------------------- layer driver --------------------------------

template <int K, int NOUT>
static void run_layer(const float* Xin, int NH, int C, int do_elu,
                      const float* W, const float* avs, const float* avd, const float* bias,
                      float* Out, _Float16* Xh, _Float16* Wt, float* Hbuf,
                      float* as_, float* ad_, float* m_, float* den_, float* alpha_e,
                      const int* ei, int N, int E, hipStream_t stream) {
  const int T = 256;
  k_transpose_f16<<<cdiv_u((long long)K * NOUT, T), T, 0, stream>>>(W, Wt, K, NOUT);
  k_f32_to_f16<<<cdiv_u((long long)N * K, T), T, 0, stream>>>(Xin, Xh, (long long)N * K);

  int totalGroups = (N / 16) * (NOUT / 64);
  k_wmma_gemm4<K, NOUT><<<cdiv_u((long long)totalGroups * 32, T), T, 0, stream>>>(
      Xh, Wt, Hbuf, totalGroups);

  k_alpha<<<cdiv_u((long long)N * NH, T), T, 0, stream>>>(Hbuf, avs, avd, as_, ad_, N, NH, C);

  k_fill_f32<<<cdiv_u((long long)N * NH, T), T, 0, stream>>>(m_, -__builtin_huge_valf(),
                                                             (long long)N * NH);
  k_fill_f32<<<cdiv_u((long long)N * NH, T), T, 0, stream>>>(den_, 0.f, (long long)N * NH);
  k_fill_f32<<<cdiv_u((long long)N * NOUT, T), T, 0, stream>>>(Out, 0.f, (long long)N * NOUT);

  long long EN = (long long)E + N;
  k_edge_max<<<cdiv_u(EN * NH, T), T, 0, stream>>>(ei, E, N, as_, ad_, m_, NH);
  k_edge_den<<<cdiv_u(EN * NH, T), T, 0, stream>>>(ei, E, N, as_, ad_, m_, den_, NH);
  k_edge_alpha<<<cdiv_u(EN * NH, T), T, 0, stream>>>(ei, E, N, as_, ad_, m_, den_, alpha_e, NH);
  k_edge_acc4<<<cdiv_u(EN * (NOUT / 4), T), T, 0, stream>>>(ei, E, N, alpha_e, Hbuf, Out,
                                                            NH, C);
  k_bias_act<<<cdiv_u((long long)N * NOUT, T), T, 0, stream>>>(Out, bias, (long long)N * NOUT,
                                                               NOUT, do_elu);
}

// ----------------------------- entry point ---------------------------------

extern "C" void kernel_launch(void* const* d_in, const int* in_sizes, int n_in,
                              void* d_out, int out_size, void* d_ws, size_t ws_size,
                              hipStream_t stream) {
  const float* x     = (const float*)d_in[0];
  const int*   ei    = (const int*)d_in[1];
  const int*   batch = (const int*)d_in[2];
  const float* W1 = (const float*)d_in[3];
  const float* aS1 = (const float*)d_in[4];
  const float* aD1 = (const float*)d_in[5];
  const float* b1 = (const float*)d_in[6];
  const float* W2 = (const float*)d_in[7];
  const float* aS2 = (const float*)d_in[8];
  const float* aD2 = (const float*)d_in[9];
  const float* b2 = (const float*)d_in[10];
  const float* W3 = (const float*)d_in[11];
  const float* aS3 = (const float*)d_in[12];
  const float* aD3 = (const float*)d_in[13];
  const float* b3 = (const float*)d_in[14];
  const float* W4 = (const float*)d_in[15];
  const float* aS4 = (const float*)d_in[16];
  const float* aD4 = (const float*)d_in[17];
  const float* b4 = (const float*)d_in[18];
  const float* Wl = (const float*)d_in[19];
  const float* bl = (const float*)d_in[20];

  const int N = in_sizes[0] / 128;   // 50000
  const int E = in_sizes[1] / 2;     // 400000
  const int B = out_size;            // 50
  const int HC = 256, FO = 64;

  // carve workspace (each chunk 256B aligned)
  char* ws = (char*)d_ws;
  size_t off = 0;
  auto carve = [&](size_t bytes) -> void* {
    void* p = ws + off;
    off = (off + bytes + 255) & ~(size_t)255;
    return p;
  };
  float*    bufA    = (float*)carve((size_t)N * HC * 4);
  float*    bufB    = (float*)carve((size_t)N * HC * 4);
  float*    Hbuf    = (float*)carve((size_t)N * HC * 4);
  _Float16* Xh      = (_Float16*)carve((size_t)N * HC * 2);
  _Float16* Wt      = (_Float16*)carve((size_t)HC * HC * 2);
  float*    as_     = (float*)carve((size_t)N * 8 * 4);
  float*    ad_     = (float*)carve((size_t)N * 8 * 4);
  float*    m_      = (float*)carve((size_t)N * 8 * 4);
  float*    den_    = (float*)carve((size_t)N * 8 * 4);
  float*    alpha_e = (float*)carve((size_t)(E + N) * 8 * 4);
  float*    pooled  = (float*)carve((size_t)B * FO * 4);
  float*    counts  = (float*)carve((size_t)B * 4);
  (void)ws_size; (void)n_in;

  // 4 GAT layers
  run_layer<128, 256>(x,    8,  32, 1, W1, aS1, aD1, b1, bufA, Xh, Wt, Hbuf,
                      as_, ad_, m_, den_, alpha_e, ei, N, E, stream);
  run_layer<256, 256>(bufA, 1, 256, 1, W2, aS2, aD2, b2, bufB, Xh, Wt, Hbuf,
                      as_, ad_, m_, den_, alpha_e, ei, N, E, stream);
  run_layer<256, 256>(bufB, 1, 256, 1, W3, aS3, aD3, b3, bufA, Xh, Wt, Hbuf,
                      as_, ad_, m_, den_, alpha_e, ei, N, E, stream);
  run_layer<256, 64>(bufA,  1,  64, 0, W4, aS4, aD4, b4, bufB, Xh, Wt, Hbuf,
                     as_, ad_, m_, den_, alpha_e, ei, N, E, stream);

  // pooling + linear head
  const int T = 256;
  k_fill_f32<<<cdiv_u((long long)B * FO, T), T, 0, stream>>>(pooled, 0.f, (long long)B * FO);
  k_fill_f32<<<1, T, 0, stream>>>(counts, 0.f, (long long)B);
  k_pool<<<cdiv_u((long long)N * FO, T), T, 0, stream>>>(bufB, batch, pooled, counts, N, FO);
  k_final<<<1, 256, 0, stream>>>(pooled, counts, Wl, bl, (float*)d_out, B, FO);
}
